// DeepSet_23304492548148
// MI455X (gfx1250) — matrix operand order repaired and checked by reading, not compiled, not run on hardware
//
#include <hip/hip_runtime.h>

typedef __attribute__((ext_vector_type(16))) _Float16 v16h;
typedef __attribute__((ext_vector_type(8)))  _Float16 v8h;
typedef __attribute__((ext_vector_type(8)))  float    v8f;

#define WMMA_F16(a, b, c) \
  __builtin_amdgcn_wmma_f32_16x16x32_f16(false, (a), false, (b), (short)0, (c), false, false)

// ---------------------------------------------------------------------------
// Pack f32 [K=64][N=64] weight matrices into the per-lane f16 B-operand layout
// for V_WMMA_F32_16X16X32_F16:
//   packed[((t*2+kb)*32 + lane)*16 + i] = W[(kb*32 + (i&7) + 8*(lane>>4) + 16*(i>>3)) * 64 + (t*16 + (lane&15))]
// Three matrices: phi_w1 @ 0, phi_w2 @ 4096, rho_w1 @ 8192 (halfs).
// ---------------------------------------------------------------------------
__global__ void pack_kernel(const float* __restrict__ w1,
                            const float* __restrict__ w2,
                            const float* __restrict__ rw1,
                            _Float16* __restrict__ pw) {
  int p = blockIdx.x * blockDim.x + threadIdx.x;   // 0 .. 12287
  if (p >= 3 * 4096) return;
  int which = p >> 12;
  int q     = p & 4095;
  int i     = q & 15;
  int lane  = (q >> 4) & 31;
  int blk   = q >> 9;          // t*2 + kb, 0..7
  int kb    = blk & 1;
  int t     = blk >> 1;
  int k     = kb * 32 + (i & 7) + ((lane >> 4) << 3) + ((i >> 3) << 4);
  int n     = t * 16 + (lane & 15);
  const float* W = (which == 0) ? w1 : ((which == 1) ? w2 : rw1);
  pw[p] = (_Float16)W[k * 64 + n];
}

__global__ void zero_kernel(float4* __restrict__ p, int n4) {
  int i = blockIdx.x * blockDim.x + threadIdx.x;
  if (i < n4) p[i] = make_float4(0.f, 0.f, 0.f, 0.f);
}

// Load 16x32 f16 A-operand slice for one lane from a row-major f32 row.
// Lane holds row M = lane&15; K runs [kOff+8*hi, +8) and [kOff+16+8*hi, +8).
__device__ __forceinline__ v16h load_a_row(const float* __restrict__ row, int hi, int kOff) {
  v16h a;
  const float* s0 = row + kOff + 8 * hi;
  const float* s1 = s0 + 16;
#pragma unroll
  for (int i = 0; i < 8; ++i) {
    a[i]     = (_Float16)s0[i];
    a[i + 8] = (_Float16)s1[i];
  }
  return a;
}

__device__ __forceinline__ void atomic_add_f32(float* p, float v) {
  unsafeAtomicAdd(p, v);  // global_atomic_add_f32 (no CAS loop)
}

// ---------------------------------------------------------------------------
// Kernel 1: persistent phi MLP (64 -> relu 64 -> 64) over 16-row tiles +
// run-compressed segment scatter-add into summ[agent][64].
// One wave per tile per iteration; all 16 WMMA B-operands (128 VGPRs) and the
// per-lane biases are hoisted into registers outside the grid-stride loop.
// ---------------------------------------------------------------------------
__global__ __launch_bounds__(256) void phi_scatter_kernel(
    const float* __restrict__ nb, const float* __restrict__ pb1,
    const float* __restrict__ pb2, const int* __restrict__ seg,
    const _Float16* __restrict__ pw, float* __restrict__ summ,
    int ntiles, int tile_stride) {
  __shared__ _Float16 lh[8][16][64];  // per-wave 16x64 f16 tile (16 KB)
  const int lane   = threadIdx.x & 31;
  const int wave   = threadIdx.x >> 5;
  const int laneLo = lane & 15;
  const int hi     = lane >> 4;
  const int wgid   = blockIdx.x * 8 + wave;

  // Hoist weights + biases: resident for the whole kernel.
  v16h B1[4][2], B2[4][2];
  float bias1[4], bias2[4];
#pragma unroll
  for (int t = 0; t < 4; ++t) {
#pragma unroll
    for (int kb = 0; kb < 2; ++kb) {
      B1[t][kb] = *(const v16h*)(pw + ((size_t)(t * 2 + kb) * 32 + lane) * 16);
      B2[t][kb] = *(const v16h*)(pw + 4096 + ((size_t)(t * 2 + kb) * 32 + lane) * 16);
    }
    bias1[t] = pb1[t * 16 + laneLo];
    bias2[t] = pb2[t * 16 + laneLo];
  }

  for (int tile = wgid; tile < ntiles; tile += tile_stride) {
    const int r0 = tile * 16;
    const float* arow = nb + (size_t)(r0 + laneLo) * 64;
    v16h a0 = load_a_row(arow, hi, 0);    // K 0..31
    v16h a1 = load_a_row(arow, hi, 32);   // K 32..63

    // ---- layer 1: 8 WMMAs, bias + ReLU, stash f16 tile in LDS ----
#pragma unroll
    for (int t = 0; t < 4; ++t) {
      v8f c = {0.f, 0.f, 0.f, 0.f, 0.f, 0.f, 0.f, 0.f};
      c = WMMA_F16(a0, B1[t][0], c);
      c = WMMA_F16(a1, B1[t][1], c);
      const int n = t * 16 + laneLo;       // N is per-lane constant in D layout
#pragma unroll
      for (int r = 0; r < 8; ++r) {
        float v = c[r] + bias1[t];
        v = v > 0.f ? v : 0.f;             // ReLU
        lh[wave][r + 8 * hi][n] = (_Float16)v;
      }
    }

    // Segment ids for this wave-half's 8 rows (two b128 loads).
    const int rowbase = r0 + 8 * hi;
    const int4 s03 = *(const int4*)(seg + rowbase);
    const int4 s47 = *(const int4*)(seg + rowbase + 4);
    int segs[8] = {s03.x, s03.y, s03.z, s03.w, s47.x, s47.y, s47.z, s47.w};

    // ---- re-layout D -> A through wave-private LDS (same-wave DS in-order) ----
    const _Float16* hrow = &lh[wave][laneLo][0];
    v8h l0 = *(const v8h*)(hrow + 8 * hi);
    v8h h0 = *(const v8h*)(hrow + 16 + 8 * hi);
    v8h l1 = *(const v8h*)(hrow + 32 + 8 * hi);
    v8h h1 = *(const v8h*)(hrow + 48 + 8 * hi);
    v16h a2 = __builtin_shufflevector(l0, h0, 0,1,2,3,4,5,6,7,8,9,10,11,12,13,14,15);
    v16h a3 = __builtin_shufflevector(l1, h1, 0,1,2,3,4,5,6,7,8,9,10,11,12,13,14,15);

    // ---- layer 2: 8 WMMAs + run-compressed scatter ----
#pragma unroll
    for (int t = 0; t < 4; ++t) {
      v8f c = {0.f, 0.f, 0.f, 0.f, 0.f, 0.f, 0.f, 0.f};
      c = WMMA_F16(a2, B2[t][0], c);
      c = WMMA_F16(a3, B2[t][1], c);
      const int n = t * 16 + laneLo;
      int cs = segs[0];
      float vs = c[0] + bias2[t];
#pragma unroll
      for (int r = 1; r < 8; ++r) {
        int s = segs[r];
        float v = c[r] + bias2[t];
        if (s == cs) {
          vs += v;
        } else {
          atomic_add_f32(&summ[(size_t)cs * 64 + n], vs);
          cs = s;
          vs = v;
        }
      }
      atomic_add_f32(&summ[(size_t)cs * 64 + n], vs);
    }
  }
}

// ---------------------------------------------------------------------------
// Kernel 2: rho MLP on pooled features: relu(summ@W + b) @ W2 + b2 -> [N,2].
// WMMA for the 64x64, VALU dot for the 64->2 tail. 5 waves (160 thr) / block.
// ---------------------------------------------------------------------------
__global__ __launch_bounds__(160) void rho_kernel(
    const float* __restrict__ summ, const _Float16* __restrict__ pwr,
    const float* __restrict__ rb1, const float* __restrict__ rw2,
    const float* __restrict__ rb2, float* __restrict__ out, int ntiles) {
  __shared__ float lr[5][16][64];  // 20 KB
  const int lane   = threadIdx.x & 31;
  const int wave   = threadIdx.x >> 5;
  const int tile   = blockIdx.x * 5 + wave;
  const bool active = tile < ntiles;   // wave-uniform
  const int laneLo = lane & 15;
  const int hi     = lane >> 4;
  const int r0     = tile * 16;

  if (active) {
    const float* arow = summ + (size_t)(r0 + laneLo) * 64;
    v16h a0 = load_a_row(arow, hi, 0);
    v16h a1 = load_a_row(arow, hi, 32);
#pragma unroll
    for (int t = 0; t < 4; ++t) {
      v16h b0 = *(const v16h*)(pwr + ((size_t)(t * 2 + 0) * 32 + lane) * 16);
      v16h b1 = *(const v16h*)(pwr + ((size_t)(t * 2 + 1) * 32 + lane) * 16);
      v8f c = {0.f, 0.f, 0.f, 0.f, 0.f, 0.f, 0.f, 0.f};
      c = WMMA_F16(a0, b0, c);
      c = WMMA_F16(a1, b1, c);
      const int n = t * 16 + laneLo;
      const float bias = rb1[n];
#pragma unroll
      for (int r = 0; r < 8; ++r) {
        float v = c[r] + bias;
        lr[wave][r + 8 * hi][n] = v > 0.f ? v : 0.f;
      }
    }
  }
  __syncthreads();
  if (active) {
    const int m = laneLo;      // agent row within tile
    const int a = hi;          // output dim 0/1
    const float* hrow = &lr[wave][m][0];
    float o = 0.f;
#pragma unroll
    for (int j = 0; j < 64; ++j) {
      int jj = (j + m) & 63;   // rotate to avoid 32-way LDS bank conflicts
      o += hrow[jj] * rw2[jj * 2 + a];
    }
    out[(size_t)(r0 + m) * 2 + a] = o + rb2[a];
  }
}

// ---------------------------------------------------------------------------
extern "C" void kernel_launch(void* const* d_in, const int* in_sizes, int n_in,
                              void* d_out, int out_size, void* d_ws, size_t ws_size,
                              hipStream_t stream) {
  const float* nb  = (const float*)d_in[0];   // neighbors [NN, 64]
  const float* w1  = (const float*)d_in[1];   // phi_w1 [64,64]
  const float* b1  = (const float*)d_in[2];   // phi_b1 [64]
  const float* w2  = (const float*)d_in[3];   // phi_w2 [64,64]
  const float* b2  = (const float*)d_in[4];   // phi_b2 [64]
  const float* rw1 = (const float*)d_in[5];   // rho_w1 [64,64]
  const float* rb1 = (const float*)d_in[6];   // rho_b1 [64]
  const float* rw2 = (const float*)d_in[7];   // rho_w2 [64,2]
  const float* rb2 = (const float*)d_in[8];   // rho_b2 [2]
  const int*   seg = (const int*)d_in[9];     // segment_ids [NN], sorted
  float* out = (float*)d_out;

  const int nn  = in_sizes[0] / 64;   // 1,600,000 neighbors
  const int nag = out_size / 2;       // 50,000 agents

  // Workspace layout: packed f16 weights (24 KB), then summ f32 [nag][64].
  _Float16* pw  = (_Float16*)d_ws;
  float*   summ = (float*)((char*)d_ws + 32768);
  (void)ws_size; (void)n_in;

  pack_kernel<<<48, 256, 0, stream>>>(w1, w2, rw1, pw);

  const int n4 = (nag * 64) / 4;
  zero_kernel<<<(n4 + 255) / 256, 256, 0, stream>>>((float4*)summ, n4);

  // Persistent phi kernel: 1024 blocks x 8 waves = 8192 waves; each wave
  // strides over ~12 tiles with weights held in registers.
  const int t1 = (nn + 15) / 16;      // 100,000 tiles
  const int blocks1 = 1024;
  phi_scatter_kernel<<<blocks1, 256, 0, stream>>>(nb, b1, b2, seg, pw, summ,
                                                  t1, blocks1 * 8);

  const int t2 = (nag + 15) / 16;     // 3,125 tiles
  rho_kernel<<<(t2 + 4) / 5, 160, 0, stream>>>(summ, pw + 8192, rb1, rw2, rb2, out, t2);
}
